// ContentAwareMambaFilter_16887811408339
// MI455X (gfx1250) — matrix operand order repaired and checked by reading, not compiled, not run on hardware
//
#include <hip/hip_runtime.h>
#include <hip/hip_bf16.h>
#include <math.h>

// ---------------------------------------------------------------------------
// ContentAwareMambaFilter for MI455X (gfx1250, wave32, WMMA + TDM).
// Five GEMMs via one branch-free bf16 WMMA kernel (v_wmma_f32_16x16x32_bf16,
// f32 accumulate), LDS double-buffered; tiles are DMA'd into LDS with
// TENSOR_LOAD_TO_LDS when the builtin is available (fallback: b128 staging).
// Scan / conv / norm stay fp32 vector code.
// ---------------------------------------------------------------------------

typedef __bf16 bf16_t;
typedef __attribute__((ext_vector_type(16))) __bf16 v16bf;
typedef __attribute__((ext_vector_type(8)))  float  v8f;
typedef __attribute__((ext_vector_type(4)))  unsigned int v4u;
typedef __attribute__((ext_vector_type(8)))  int v8i;
typedef __attribute__((ext_vector_type(4)))  int v4i;

union FragBF { v16bf v; v4u u[2]; };

#define DIM_    768
#define DINNER_ 1536
#define DSTATE_ 16
#define DTRANK_ 48
#define DTRANKP_ 64        // dt_proj K padded 48 -> 64
#define DCONV_  4
#define BB_     8
#define LL_     2048
#define TT_     (BB_*LL_)
#define PROJW_  (DTRANK_ + 2*DSTATE_)   // 80 (logical)
#define PROJP_  128                     // proj buffer stride, N padded 80 -> 128

// ---- CDNA5 Tensor Data Mover availability / arity selection ----
#if defined(__has_builtin)
#  if __has_builtin(__builtin_amdgcn_tensor_load_to_lds)
#    define USE_TDM 1
#  else
#    define USE_TDM 0
#  endif
#else
#  define USE_TDM 0
#endif
#if defined(__has_include)
#  if __has_include(<hip/amd_detail/amd_gfx1250_TDM.h>)
#    define TDM_ARGS6 1   // therock/clang-23 headers -> 6-arg builtin
#  else
#    define TDM_ARGS6 0   // ROCm 7.2 / clang-22 -> 5-arg builtin
#  endif
#else
#  define TDM_ARGS6 0
#endif

__device__ __forceinline__ float sigmoidf_(float v){ return 1.f/(1.f+__expf(-v)); }

#if USE_TDM
// 2D tile load: rows x 64 bytes (32 bf16) from a row-major matrix with
// row_bytes stride, into LDS rows padded to 80 bytes.
// D# per CDNA5 ISA: data_size=1B; pad_interval=16 DWORDs
// (code 3), pad_amount=4 DWORDs (code 3); group0.type=2, count=1.
__device__ __forceinline__ void tdm_tile(const bf16_t* g, unsigned lds_off,
                                         unsigned row_bytes, unsigned rows){
  unsigned long long ga = (unsigned long long)(size_t)g;
  v4u g0;
  g0.x = 1u;                                                  // count=1
  g0.y = lds_off;                                             // lds_addr
  g0.z = (unsigned)ga;                                        // global_addr[31:0]
  g0.w = (unsigned)((ga >> 32) & 0x01FFFFFFu) | (2u << 30);   // ga[56:32] | type=2
  v8i g1;
  g1[0] = (int)((1u << 20) | (3u << 22) | (3u << 25));        // pad_en|interval|amount
  g1[1] = (int)((row_bytes & 0xFFFFu) << 16);                 // tensor_dim0[15:0]
  g1[2] = (int)((row_bytes >> 16) | ((rows & 0xFFFFu) << 16));// dim0[31:16]|dim1[15:0]
  g1[3] = (int)(((rows >> 16) & 0xFFFFu) | (64u << 16));      // dim1[31:16]|tile_dim0=64B
  g1[4] = (int)(rows & 0xFFFFu);                              // tile_dim1 (tile_dim2=0)
  g1[5] = (int)row_bytes;                                     // tensor_dim0_stride[31:0]
  g1[6] = 0;                                                  // stride0[47:32], stride1 lo
  g1[7] = 0;
  v4i z4 = {0,0,0,0};
#if TDM_ARGS6
  v8i z8 = {0,0,0,0,0,0,0,0};
  __builtin_amdgcn_tensor_load_to_lds(g0, g1, z4, z4, z8, 0);
#else
  __builtin_amdgcn_tensor_load_to_lds(g0, g1, z4, z4, 0);
#endif
}
#endif // USE_TDM

// ---------------- weight convert + transpose:  wt[n*Kp + k] = w[k*N + n] ----
__global__ void k_wtrans(const float* __restrict__ w, bf16_t* __restrict__ wt,
                         int K, int N, int Kp, int Np){
  int i = blockIdx.x*blockDim.x + threadIdx.x;     // Np*Kp
  if (i >= Np*Kp) return;
  int n = i / Kp, k = i % Kp;
  wt[i] = (n < N && k < K) ? (bf16_t)w[(size_t)k*N + n] : (bf16_t)0.f;
}

// h = relu(saliency * w1 + b1), bf16 (T x 192)
__global__ void k_h(const float* __restrict__ sal, const float* __restrict__ w1,
                    const float* __restrict__ b1, bf16_t* __restrict__ h){
  int i = blockIdx.x*blockDim.x + threadIdx.x;      // T*192
  int t = i / (DIM_/4); int j = i % (DIM_/4);
  float v = fmaf(sal[t], w1[j], b1[j]);
  h[i] = (bf16_t)fmaxf(v, 0.f);
}

// x_mod = x*(1+tanh(gamma)) + beta, bf16 (T x 768)
__global__ void k_xmod(const float* __restrict__ x, const float* __restrict__ affine,
                       bf16_t* __restrict__ xm){
  int i = blockIdx.x*blockDim.x + threadIdx.x;      // T*768
  int t = i / DIM_; int j = i % DIM_;
  float g  = affine[(size_t)t*(2*DIM_) + j];
  float be = affine[(size_t)t*(2*DIM_) + DIM_ + j];
  xm[i] = (bf16_t)(x[i]*(1.f + tanhf(g)) + be);
}

// causal depthwise conv (K=4) + SiLU; writes xc f32 (scan) and xc_bf (GEMM-A)
__global__ void k_conv(const bf16_t* __restrict__ xz, const float* __restrict__ cw,
                       const float* __restrict__ cb, float* __restrict__ xc,
                       bf16_t* __restrict__ xcb){
  int i = blockIdx.x*blockDim.x + threadIdx.x;      // T*1536
  int d = i % DINNER_; int t = i / DINNER_; int l = t % LL_;
  float acc = cb[d];
  #pragma unroll
  for (int k = 0; k < DCONV_; ++k){
    int ll = l - (DCONV_-1) + k;
    float xv = (ll >= 0) ? (float)xz[(size_t)(t - (DCONV_-1) + k)*(2*DINNER_) + d] : 0.f;
    acc = fmaf(cw[d*DCONV_ + k], xv, acc);
  }
  float sv = acc * sigmoidf_(acc);
  xc[i] = sv;
  xcb[i] = (bf16_t)sv;
}

// dt_in = proj[:, :48] -> bf16, K padded to 64 with zeros
__global__ void k_dtin(const float* __restrict__ proj, bf16_t* __restrict__ dtin){
  int i = blockIdx.x*blockDim.x + threadIdx.x;      // T*64
  int t = i / DTRANKP_; int j = i % DTRANKP_;
  dtin[i] = (j < DTRANK_) ? (bf16_t)proj[(size_t)t*PROJP_ + j] : (bf16_t)0.f;
}

// selective scan: one thread per (b, d); 16 states in registers.
// Fuses y = (scan + D*xc) * silu(z), emits bf16 GEMM-A for out_proj.
__global__ void k_scan(const float* __restrict__ dt, const float* __restrict__ xc,
                       const float* __restrict__ proj, const bf16_t* __restrict__ xz,
                       const float* __restrict__ A_log, const float* __restrict__ Dp,
                       bf16_t* __restrict__ ybf){
  int i = blockIdx.x*blockDim.x + threadIdx.x;      // B*DINNER = 12288
  int d = i % DINNER_; int b = i / DINNER_;
  float Arow[DSTATE_], h[DSTATE_];
  #pragma unroll
  for (int n = 0; n < DSTATE_; ++n){
    Arow[n] = -__expf(A_log[(size_t)d*DSTATE_ + n]);
    h[n] = 0.f;
  }
  float Dd = Dp[d];
  size_t base = (size_t)b * LL_;
  for (int t = 0; t < LL_; ++t){
    size_t row = base + t;
    float dtv = dt[row*DINNER_ + d];
    float xv  = xc[row*DINNER_ + d];
    const float* pr = proj + row*PROJP_;            // broadcast across wave lanes
    float dx = dtv * xv;
    float y = 0.f;
    #pragma unroll
    for (int n = 0; n < DSTATE_; ++n){
      float hn = fmaf(h[n], __expf(dtv*Arow[n]), dx * pr[DTRANK_ + n]);
      h[n] = hn;
      y = fmaf(hn, pr[DTRANK_ + DSTATE_ + n], y);
    }
    float zv = (float)xz[row*(2*DINNER_) + DINNER_ + d];
    float gate = zv * sigmoidf_(zv);
    ybf[row*DINNER_ + d] = (bf16_t)((y + Dd*xv) * gate);
  }
}

// r = x + 0.1*x_mamba; LayerNorm over 768; one block (256 thr) per token.
__global__ void k_lnorm(const float* __restrict__ x, const float* __restrict__ xm,
                        const float* __restrict__ g, const float* __restrict__ bb,
                        float* __restrict__ out){
  int t = blockIdx.x; int tid = threadIdx.x;
  __shared__ float red[256];
  float v[3]; float s = 0.f;
  #pragma unroll
  for (int i = 0; i < 3; ++i){
    int j = tid + i*256;
    float r = x[(size_t)t*DIM_ + j] + 0.1f*xm[(size_t)t*DIM_ + j];
    v[i] = r; s += r;
  }
  red[tid] = s; __syncthreads();
  for (int off = 128; off > 0; off >>= 1){
    if (tid < off) red[tid] += red[tid + off];
    __syncthreads();
  }
  float mu = red[0] * (1.f/DIM_);
  __syncthreads();
  float q = 0.f;
  #pragma unroll
  for (int i = 0; i < 3; ++i){ float dd = v[i] - mu; q = fmaf(dd, dd, q); }
  red[tid] = q; __syncthreads();
  for (int off = 128; off > 0; off >>= 1){
    if (tid < off) red[tid] += red[tid + off];
    __syncthreads();
  }
  float inv = rsqrtf(red[0] * (1.f/DIM_) + 1e-5f);
  #pragma unroll
  for (int i = 0; i < 3; ++i){
    int j = tid + i*256;
    out[(size_t)t*DIM_ + j] = (v[i] - mu) * inv * g[j] + bb[j];
  }
}

// ---------------- bf16 WMMA GEMM (double-buffered LDS, TDM staged) ---------
// C[MxN] = act(A[MxK] @ B[KxN] + bias), B given TRANSPOSED: BT[N][K].
// Requires M%64==0, N%128==0, K%32==0 (guaranteed via padded buffers) ->
// branch structure is wave-uniform; EXEC all-ones at every WMMA.
// Workgroup 256 thr (8 waves); tile 64(M) x 128(N), BK=32, 2 LDS buffers.
// Wave w: mt=w&3 (16-row M subtile), nt=w>>2 (64-col N half); 4 WMMA/wave/step.
// Fragments (CDNA5 ISA 7.12.2, wave32), r=lane&15, hl=lane>>4:
//   A row M=r:  e<8 -> K=8*hl+e ; e>=8 -> K=16+8*hl+(e-8)   => 2 x b128 LDS
//   B col N=r:  e<8 -> K=8*hl+e ; e>=8 -> K=16+8*hl+(e-8)   => 2 x b128 LDS
//   C/D: element i -> M = i + 8*hl, N=r
__global__ __launch_bounds__(256) void k_gemm_bf16(
    const bf16_t* __restrict__ A, const bf16_t* __restrict__ BT,
    float* __restrict__ Cf, bf16_t* __restrict__ Cb, const float* __restrict__ bias,
    int M, int N, int K, int act){
  __shared__ bf16_t As[2][64][40];    // 64 rows x 32 K (+8 pad -> 80B rows)
  __shared__ bf16_t Bs[2][128][40];   // 128 cols x 32 K (+8 pad)
  const int tid  = threadIdx.x;
  const int lane = tid & 31, r = lane & 15, hl = lane >> 4;
  const int w = tid >> 5, mt = w & 3, nt = w >> 2;
  const int m0 = blockIdx.y * 64, n0 = blockIdx.x * 128;

  v8f zero = {};
  v8f acc[4] = { zero, zero, zero, zero };

  // per-thread staging coordinates (fallback path)
  const int aR = tid >> 2,        aC = (tid & 3) << 3;
  const int bR0 = (tid*2) >> 2,   bC0 = ((tid*2) & 3) << 3;
  const int bR1 = (tid*2+1) >> 2, bC1 = ((tid*2+1) & 3) << 3;

  // ---- preload tile 0 into buffer 0 ----
#if USE_TDM
  if (w == 0){
    tdm_tile(&A [(size_t)m0*K], (unsigned)(size_t)&As[0][0][0], (unsigned)K*2u,  64u);
    tdm_tile(&BT[(size_t)n0*K], (unsigned)(size_t)&Bs[0][0][0], (unsigned)K*2u, 128u);
    __builtin_amdgcn_s_wait_tensorcnt(0);
  }
#else
  *(v4u*)&As[0][aR ][aC ] = *(const v4u*)&A [(size_t)(m0 + aR )*K + aC ];
  *(v4u*)&Bs[0][bR0][bC0] = *(const v4u*)&BT[(size_t)(n0 + bR0)*K + bC0];
  *(v4u*)&Bs[0][bR1][bC1] = *(const v4u*)&BT[(size_t)(n0 + bR1)*K + bC1];
#endif
  __syncthreads();

  int cur = 0;
  for (int k0 = 0; k0 < K; k0 += 32){
    const int nxt = cur ^ 1;
    const bool more = (k0 + 32) < K;

    // ---- start prefetch of tile k0+32 into the other buffer ----
#if USE_TDM
    if (w == 0 && more){
      tdm_tile(&A [(size_t)m0*K + k0 + 32], (unsigned)(size_t)&As[nxt][0][0], (unsigned)K*2u,  64u);
      tdm_tile(&BT[(size_t)n0*K + k0 + 32], (unsigned)(size_t)&Bs[nxt][0][0], (unsigned)K*2u, 128u);
    }
#else
    v4u pa = {}, pb0 = {}, pb1 = {};
    if (more){
      pa  = *(const v4u*)&A [(size_t)(m0 + aR )*K + k0 + 32 + aC ];
      pb0 = *(const v4u*)&BT[(size_t)(n0 + bR0)*K + k0 + 32 + bC0];
      pb1 = *(const v4u*)&BT[(size_t)(n0 + bR1)*K + k0 + 32 + bC1];
    }
#endif

    // ---- compute on current buffer ----
    FragBF a;
    a.u[0] = *(const v4u*)&As[cur][16*mt + r][ 8*hl];
    a.u[1] = *(const v4u*)&As[cur][16*mt + r][16 + 8*hl];
    #pragma unroll
    for (int nn = 0; nn < 4; ++nn){
      FragBF b;
      const bf16_t* bp = &Bs[cur][64*nt + 16*nn + r][8*hl];
      b.u[0] = *(const v4u*)(bp);
      b.u[1] = *(const v4u*)(bp + 16);
      acc[nn] = __builtin_amdgcn_wmma_f32_16x16x32_bf16(false, a.v, false, b.v,
                                                        (short)0, acc[nn], false, false);
    }

    // ---- finish prefetch, hand off buffers ----
#if USE_TDM
    if (w == 0) __builtin_amdgcn_s_wait_tensorcnt(0);
#else
    if (more){
      *(v4u*)&As[nxt][aR ][aC ] = pa;
      *(v4u*)&Bs[nxt][bR0][bC0] = pb0;
      *(v4u*)&Bs[nxt][bR1][bC1] = pb1;
    }
#endif
    __syncthreads();
    cur = nxt;
  }

  #pragma unroll
  for (int nn = 0; nn < 4; ++nn){
    int col = n0 + 64*nt + 16*nn + r;
    float bv = bias ? bias[col] : 0.f;
    #pragma unroll
    for (int i = 0; i < 8; ++i){
      int row = m0 + 16*mt + i + 8*hl;
      float v = acc[nn][i] + bv;
      if (act == 1) v = (v > 20.f) ? v : log1pf(__expf(v));   // softplus
      if (Cf) Cf[(size_t)row*N + col] = v;
      else    Cb[(size_t)row*N + col] = (bf16_t)v;
    }
  }
}

// ---------------- launch ----------------

extern "C" void kernel_launch(void* const* d_in, const int* in_sizes, int n_in,
                              void* d_out, int out_size, void* d_ws, size_t ws_size,
                              hipStream_t stream){
  (void)in_sizes; (void)n_in; (void)out_size; (void)ws_size;

  const float* x        = (const float*)d_in[0];
  const float* sal      = (const float*)d_in[1];
  const float* sp_w1    = (const float*)d_in[2];
  const float* sp_b1    = (const float*)d_in[3];
  const float* sp_w2    = (const float*)d_in[4];
  const float* sp_b2    = (const float*)d_in[5];
  const float* in_proj  = (const float*)d_in[6];
  const float* conv_w   = (const float*)d_in[7];
  const float* conv_b   = (const float*)d_in[8];
  const float* x_proj   = (const float*)d_in[9];
  const float* dt_proj  = (const float*)d_in[10];
  const float* dt_bias  = (const float*)d_in[11];
  const float* A_log    = (const float*)d_in[12];
  const float* Dv       = (const float*)d_in[13];
  const float* out_proj = (const float*)d_in[14];
  const float* ln_g     = (const float*)d_in[15];
  const float* ln_b     = (const float*)d_in[16];

  // ---- workspace arena (~370 MB, regions reused along the dependence chain)
  char* base = (char*)d_ws;
  size_t off = 0;
  auto arena = [&](size_t bytes)->void*{
    void* p = base + off;
    off += (bytes + 255) & ~(size_t)255;
    return p;
  };
  bf16_t* xz_bf   = (bf16_t*)arena((size_t)TT_*2*DINNER_*sizeof(bf16_t)); // xin|z
  float*  affine  = (float*) arena((size_t)TT_*2*DIM_*sizeof(float));     // -> dt
  float*  dtbuf   = affine;                                               // reuse
  float*  xc      = (float*) arena((size_t)TT_*DINNER_*sizeof(float));    // -> xmamba
  float*  xmamba  = xc;                                                   // reuse
  bf16_t* scr     = (bf16_t*)arena((size_t)TT_*DINNER_*sizeof(bf16_t));   // h/xmod/xc_bf/y_bf
  float*  proj    = (float*) arena((size_t)TT_*PROJP_*sizeof(float));     // stride 128
  bf16_t* dtin_bf = (bf16_t*)arena((size_t)TT_*DTRANKP_*sizeof(bf16_t));  // K padded 64
  // transposed bf16 weights: wt[N][K]
  bf16_t* w2bT    = (bf16_t*)arena((size_t)(2*DIM_)*(DIM_/4)*sizeof(bf16_t));   // 1536 x 192
  bf16_t* inwT    = (bf16_t*)arena((size_t)(2*DINNER_)*DIM_*sizeof(bf16_t));    // 3072 x 768
  bf16_t* xpwT    = (bf16_t*)arena((size_t)PROJP_*DINNER_*sizeof(bf16_t));      // 128 x 1536
  bf16_t* dtwT    = (bf16_t*)arena((size_t)DINNER_*DTRANKP_*sizeof(bf16_t));    // 1536 x 64
  bf16_t* opwT    = (bf16_t*)arena((size_t)DIM_*DINNER_*sizeof(bf16_t));        // 768 x 1536

  dim3 blk(256);
  auto g1 = [](size_t n){ return dim3((unsigned)((n + 255)/256)); };

  // weight transposes (f32 [K][N] -> bf16 [Np][Kp], zero padded)
  k_wtrans<<<g1((size_t)(2*DIM_)*(DIM_/4)),   blk, 0, stream>>>(sp_w2,    w2bT, DIM_/4,  2*DIM_,    DIM_/4,   2*DIM_);
  k_wtrans<<<g1((size_t)(2*DINNER_)*DIM_),    blk, 0, stream>>>(in_proj,  inwT, DIM_,    2*DINNER_, DIM_,     2*DINNER_);
  k_wtrans<<<g1((size_t)PROJP_*DINNER_),      blk, 0, stream>>>(x_proj,   xpwT, DINNER_, PROJW_,    DINNER_,  PROJP_);
  k_wtrans<<<g1((size_t)DINNER_*DTRANKP_),    blk, 0, stream>>>(dt_proj,  dtwT, DTRANK_, DINNER_,   DTRANKP_, DINNER_);
  k_wtrans<<<g1((size_t)DIM_*DINNER_),        blk, 0, stream>>>(out_proj, opwT, DINNER_, DIM_,      DINNER_,  DIM_);

  // saliency MLP + FiLM
  k_h<<<g1((size_t)TT_*(DIM_/4)), blk, 0, stream>>>(sal, sp_w1, sp_b1, scr);
  {
    dim3 g((2*DIM_)/128, TT_/64);
    k_gemm_bf16<<<g, blk, 0, stream>>>(scr, w2bT, affine, nullptr, sp_b2,
                                       TT_, 2*DIM_, DIM_/4, 0);
  }
  k_xmod<<<g1((size_t)TT_*DIM_), blk, 0, stream>>>(x, affine, scr);

  // in_proj: (T x 768) @ (768 x 3072) -> bf16 xz
  {
    dim3 g((2*DINNER_)/128, TT_/64);
    k_gemm_bf16<<<g, blk, 0, stream>>>(scr, inwT, nullptr, xz_bf, nullptr,
                                       TT_, 2*DINNER_, DIM_, 0);
  }

  // conv + SiLU
  k_conv<<<g1((size_t)TT_*DINNER_), blk, 0, stream>>>(xz_bf, conv_w, conv_b, xc, scr);

  // x_proj: (T x 1536) @ (1536 x 128pad) -> proj f32 (cols 80..127 = 0)
  {
    dim3 g(PROJP_/128, TT_/64);
    k_gemm_bf16<<<g, blk, 0, stream>>>(scr, xpwT, proj, nullptr, nullptr,
                                       TT_, PROJP_, DINNER_, 0);
  }

  // dt = softplus(dt_in @ dt_proj + b): (T x 64pad) @ (64pad x 1536)
  k_dtin<<<g1((size_t)TT_*DTRANKP_), blk, 0, stream>>>(proj, dtin_bf);
  {
    dim3 g(DINNER_/128, TT_/64);
    k_gemm_bf16<<<g, blk, 0, stream>>>(dtin_bf, dtwT, dtbuf, nullptr, dt_bias,
                                       TT_, DINNER_, DTRANKP_, 1);
  }

  // selective scan (fused D*xc + silu(z) gating), emits bf16 y
  k_scan<<<dim3((BB_*DINNER_)/256), blk, 0, stream>>>(dtbuf, xc, proj, xz_bf,
                                                      A_log, Dv, scr);

  // out_proj: (T x 1536) @ (1536 x 768) -> xmamba f32
  {
    dim3 g(DIM_/128, TT_/64);
    k_gemm_bf16<<<g, blk, 0, stream>>>(scr, opwT, xmamba, nullptr, nullptr,
                                       TT_, DIM_, DINNER_, 0);
  }

  // residual + LayerNorm
  k_lnorm<<<dim3(TT_), blk, 0, stream>>>(x, xmamba, ln_g, ln_b, (float*)d_out);
}
